// GQAttention_55259049230610
// MI455X (gfx1250) — matrix operand order repaired and checked by reading, not compiled, not run on hardware
//
#include <hip/hip_runtime.h>
#include <hip/hip_bf16.h>

typedef __attribute__((ext_vector_type(16))) __bf16 v16bf;
typedef __attribute__((ext_vector_type(8)))  float  v8f;

namespace gqa {

constexpr int kS   = 2048;
constexpr int kD   = 2048;
constexpr int kNH  = 16;
constexpr int kNKV = 4;
constexpr int kHD  = 128;

union BfFrag  { v16bf v; unsigned u[8]; };
union F32Frag { v8f   v; float    f[8]; };

__device__ __forceinline__ unsigned short f2bf(float x) {
  __bf16 b = (__bf16)x;
  return __builtin_bit_cast(unsigned short, b);
}

// A-fragment (16x32 bf16) dword j -> K offset, per ISA 7.12.2:
// lanes 0-15: K = {0..7, 16..23}; lanes 16-31: +8
__device__ __forceinline__ int a_k2(int j, int hi) {
  return hi * 8 + ((j < 4) ? (2 * j) : (16 + 2 * (j - 4)));
}

// Async 16B global->LDS copy (GVS mode: mem = SADDR + VADDR). ASYNCcnt-tracked.
__device__ __forceinline__ void async_copy16(unsigned lds_addr, const void* base,
                                             unsigned byte_off) {
  asm volatile("global_load_async_to_lds_b128 %0, %1, %2"
               :: "v"(lds_addr), "v"(byte_off), "s"(base)
               : "memory");
}
__device__ __forceinline__ void wait_async0() {
  asm volatile("s_wait_asynccnt 0x0" ::: "memory");
}

struct __align__(8) US4 { unsigned short x, y, z, w; };

// ---------------------------------------------------------------------------
// f32 -> bf16 bulk convert, 8 elems/thread, two 8B stores.
// ---------------------------------------------------------------------------
__global__ __launch_bounds__(256) void to_bf16(const float* __restrict__ src,
                                               unsigned short* __restrict__ dst) {
  const size_t i = ((size_t)blockIdx.x * 256 + threadIdx.x) * 8;
  float4 v0 = *(const float4*)(src + i);
  float4 v1 = *(const float4*)(src + i + 4);
  US4 a = {f2bf(v0.x), f2bf(v0.y), f2bf(v0.z), f2bf(v0.w)};
  US4 b = {f2bf(v1.x), f2bf(v1.y), f2bf(v1.z), f2bf(v1.w)};
  *(US4*)(dst + i)     = a;
  *(US4*)(dst + i + 4) = b;
}

// ---------------------------------------------------------------------------
// Y[s,n] = sum_d X[s,d] * W[n,d]   (X:[S,D] bf16, W:[N,D] bf16, Y:[S,N] f32)
// 256 threads = 8 waves, 128x64 tile, K-step 64.
// Double-buffered async global->LDS staging overlapped with WMMA compute.
// LDS row stride 72 halves = 144B: 16B-aligned rows, conflict-free fragments.
// ---------------------------------------------------------------------------
__global__ __launch_bounds__(256) void gemm_bf16(const unsigned short* __restrict__ X,
                                                 const unsigned short* __restrict__ W,
                                                 float* __restrict__ Y,
                                                 int N, int D) {
  constexpr int LDT = 72;
  __shared__ __align__(16) unsigned short lA[2][128 * LDT];
  __shared__ __align__(16) unsigned short lB[2][64 * LDT];
  const int t    = threadIdx.x;
  const int lane = t & 31;
  const int wid  = t >> 5;
  const int n16  = lane & 15;
  const int hi   = lane >> 4;
  const int wr   = wid & 3;   // M group (4 x 32 rows)
  const int wc   = wid >> 2;  // N group (2 x 32 cols)
  const int s0   = blockIdx.y * 128;
  const int n0   = blockIdx.x * 64;

  auto stage = [&](int buf, int k0) {
#pragma unroll
    for (int i = 0; i < 4; ++i) {  // A: 128x64 = 1024 16B vectors
      int idx = i * 256 + t;
      int row = idx >> 3, col = (idx & 7) * 8;
      async_copy16((unsigned)(size_t)&lA[buf][row * LDT + col], X,
                   (unsigned)(((size_t)(s0 + row) * D + k0 + col) * 2));
    }
#pragma unroll
    for (int i = 0; i < 2; ++i) {  // B: 64x64 = 512 16B vectors
      int idx = i * 256 + t;
      int row = idx >> 3, col = (idx & 7) * 8;
      async_copy16((unsigned)(size_t)&lB[buf][row * LDT + col], W,
                   (unsigned)(((size_t)(n0 + row) * D + k0 + col) * 2));
    }
  };

  F32Frag acc[2][2] = {};

  stage(0, 0);
  wait_async0();
  __syncthreads();

  const int steps = D / 64;
  for (int st = 0; st < steps; ++st) {
    const int cur = st & 1;
    if (st + 1 < steps) stage(cur ^ 1, (st + 1) * 64);  // overlap next copy

    const unsigned short* A = lA[cur];
    const unsigned short* B = lB[cur];
#pragma unroll
    for (int kc = 0; kc < 64; kc += 32) {
      BfFrag a[2], b[2];
#pragma unroll
      for (int mi = 0; mi < 2; ++mi) {
        int m = wr * 32 + mi * 16 + n16;
#pragma unroll
        for (int j = 0; j < 8; ++j)
          a[mi].u[j] = *(const unsigned*)(A + m * LDT + kc + a_k2(j, hi));
      }
#pragma unroll
      for (int ni = 0; ni < 2; ++ni) {
        int n = wc * 32 + ni * 16 + n16;
#pragma unroll
        for (int j = 0; j < 8; ++j)
          b[ni].u[j] = *(const unsigned*)(B + n * LDT + kc + a_k2(j, hi));
      }
#pragma unroll
      for (int mi = 0; mi < 2; ++mi)
#pragma unroll
        for (int ni = 0; ni < 2; ++ni)
          acc[mi][ni].v = __builtin_amdgcn_wmma_f32_16x16x32_bf16(
              false, a[mi].v, false, b[ni].v, (short)0, acc[mi][ni].v, false, false);
    }
    wait_async0();
    __syncthreads();
  }

#pragma unroll
  for (int mi = 0; mi < 2; ++mi)
#pragma unroll
    for (int ni = 0; ni < 2; ++ni)
#pragma unroll
      for (int r = 0; r < 8; ++r)
        Y[(size_t)(s0 + wr * 32 + mi * 16 + r + 8 * hi) * N +
          n0 + wc * 32 + ni * 16 + n16] = acc[mi][ni].f[r];
}

// ---------------------------------------------------------------------------
// Per-head RMSNorm + partial RoPE (64 dims) + optional gain*scale, f32 -> bf16.
// ---------------------------------------------------------------------------
__global__ __launch_bounds__(128) void normrope(const float* __restrict__ in,
                                                const float* __restrict__ gain,
                                                unsigned short* __restrict__ out,
                                                int H, float scale) {
  const int wid  = threadIdx.x >> 5;
  const int lane = threadIdx.x & 31;
  const int gid  = blockIdx.x * 4 + wid;   // gid = s*H + h
  const int s    = gid / H;
  const int h    = gid % H;
  const float* row = in + (size_t)gid * kHD;

  float4 v = *(const float4*)(row + lane * 4);
  float s2 = v.x * v.x + v.y * v.y + v.z * v.z + v.w * v.w;
#pragma unroll
  for (int off = 16; off >= 1; off >>= 1) s2 += __shfl_xor(s2, off, 32);
  const float rms = rsqrtf(s2 * (1.0f / 128.0f) + 1.1920929e-7f);
  const float g   = scale * (gain ? gain[h] : 1.0f);

  float a[4] = {v.x * rms, v.y * rms, v.z * rms, v.w * rms};
  const int e = lane * 4;
  if (e < 64) {  // rope region: pairs (i, i+32); partner lives in lane^8
    float p[4];
#pragma unroll
    for (int j = 0; j < 4; ++j) p[j] = __shfl_xor(a[j], 8, 32);
#pragma unroll
    for (int j = 0; j < 4; ++j) {
      int i = (lane < 8) ? (e + j) : (e + j - 32);
      float invf = __powf(10000.0f, -(float)i * (1.0f / 32.0f));
      float sn, cs;
      __sincosf((float)s * invf, &sn, &cs);
      a[j] = (lane < 8) ? (a[j] * cs + p[j] * sn) : (a[j] * cs - p[j] * sn);
    }
  }
  US4 o = {f2bf(a[0] * g), f2bf(a[1] * g), f2bf(a[2] * g), f2bf(a[3] * g)};
  *(US4*)(out + (size_t)gid * kHD + e) = o;
}

// ---------------------------------------------------------------------------
// V f32 [S, NKV*HD] -> Vt bf16 [NKV*HD, S]
// ---------------------------------------------------------------------------
__global__ __launch_bounds__(256) void vtrans(const float* __restrict__ V,
                                              unsigned short* __restrict__ Vt) {
  __shared__ float tile[32][33];
  const int c0 = blockIdx.x * 32;
  const int s0 = blockIdx.y * 32;
  const int tx = threadIdx.x & 31;
  const int ty = threadIdx.x >> 5;
#pragma unroll
  for (int i = 0; i < 4; ++i) {
    int r = ty + i * 8;
    tile[r][tx] = V[(size_t)(s0 + r) * (kNKV * kHD) + c0 + tx];
  }
  __syncthreads();
#pragma unroll
  for (int i = 0; i < 4; ++i) {
    int r = ty + i * 8;
    Vt[(size_t)(c0 + r) * kS + s0 + tx] = f2bf(tile[tx][r]);
  }
}

// ---------------------------------------------------------------------------
// Causal GQA flash attention, block-cooperative K/V staging.
// Block = 4 waves = 64 query rows of one head; per 32-key tile the block
// async-stages K (32x128) and V^T (128x32) into double-buffered LDS; each
// wave runs its own online softmax on a 16-row tile. Qb:[S,NH*HD] bf16,
// Kb:[S,NKV*HD] bf16, Vt:[NKV*HD,S] bf16, Ao:[S,NH*HD] bf16.
// ---------------------------------------------------------------------------
__global__ __launch_bounds__(128) void flash_attn(const unsigned short* __restrict__ Qb,
                                                  const unsigned short* __restrict__ Kb,
                                                  const unsigned short* __restrict__ Vt,
                                                  unsigned short* __restrict__ Ao) {
  constexpr int LDK = 136;  // 32 rows (kv) x 128 cols (hd), stride 272B
  constexpr int LDV = 40;   // 128 rows (hd) x 32 cols (kv), stride 80B
  __shared__ __align__(16) unsigned short lK[2][32 * LDK];
  __shared__ __align__(16) unsigned short lV[2][128 * LDV];
  __shared__ __align__(16) unsigned short lP[4][16 * 36];

  const int t    = threadIdx.x;
  const int wid  = t >> 5;
  const int lane = t & 31;
  const int n16  = lane & 15;
  const int hi   = lane >> 4;
  const int h    = blockIdx.y;
  const int qb0  = blockIdx.x * 64;
  const int q0   = qb0 + wid * 16;
  const int kvh  = h >> 2;  // NH/NKV = 4
  unsigned short* myP = lP[wid];

  auto stage = [&](int buf, int kv0) {
#pragma unroll
    for (int i = 0; i < 4; ++i) {  // K tile: 32x128 = 512 16B vectors
      int idx = i * 128 + t;
      int row = idx >> 4, col = (idx & 15) * 8;
      async_copy16((unsigned)(size_t)&lK[buf][row * LDK + col], Kb,
                   (unsigned)(((size_t)(kv0 + row) * (kNKV * kHD) + kvh * kHD + col) * 2));
    }
#pragma unroll
    for (int i = 0; i < 4; ++i) {  // V^T tile: 128x32 = 512 16B vectors
      int idx = i * 128 + t;
      int row = idx >> 2, col = (idx & 3) * 8;
      async_copy16((unsigned)(size_t)&lV[buf][row * LDV + col], Vt,
                   (unsigned)(((size_t)(kvh * kHD + row) * kS + kv0 + col) * 2));
    }
  };

  // Resident Q fragments: 4 chunks of K=32 covering HD=128 (two 16B runs each)
  BfFrag aq[4];
  {
    const unsigned short* qrow = Qb + (size_t)(q0 + n16) * (kNH * kHD) + h * kHD;
#pragma unroll
    for (int c = 0; c < 4; ++c) {
      uint4 lo = *(const uint4*)(qrow + c * 32 + hi * 8);
      uint4 hh = *(const uint4*)(qrow + c * 32 + hi * 8 + 16);
      aq[c].u[0] = lo.x; aq[c].u[1] = lo.y; aq[c].u[2] = lo.z; aq[c].u[3] = lo.w;
      aq[c].u[4] = hh.x; aq[c].u[5] = hh.y; aq[c].u[6] = hh.z; aq[c].u[7] = hh.w;
    }
  }

  F32Frag O[8] = {};
  float M[8], L[8];
#pragma unroll
  for (int r = 0; r < 8; ++r) { M[r] = -3.0e38f; L[r] = 0.f; }

  const int nT = 2 * blockIdx.x + 2;  // block-uniform tile count
  stage(0, 0);
  wait_async0();
  __syncthreads();

  for (int tile = 0; tile < nT; ++tile) {
    const int kv0 = tile * 32;
    const int cur = tile & 1;
    if (tile + 1 < nT) stage(cur ^ 1, kv0 + 32);  // overlap next copy

    if (kv0 < q0 + 16) {  // wave-uniform causal bound: EXEC stays all-ones
      const unsigned short* K = lK[cur];
      const unsigned short* V = lV[cur];

      // ---- scores S = Q K^T for 32 keys (two 16-col tiles) ----
      F32Frag sc[2];
#pragma unroll
      for (int nt = 0; nt < 2; ++nt) {
        v8f c = {};
#pragma unroll
        for (int ck = 0; ck < 4; ++ck) {
          BfFrag b;
#pragma unroll
          for (int j = 0; j < 8; ++j)
            b.u[j] = *(const unsigned*)(K + (nt * 16 + n16) * LDK + ck * 32 + a_k2(j, hi));
          c = __builtin_amdgcn_wmma_f32_16x16x32_bf16(false, aq[ck].v, false, b.v,
                                                      (short)0, c, false, false);
        }
        sc[nt].v = c;
      }

      // ---- causal mask (C layout: element (r + 8*hi, n16)) ----
#pragma unroll
      for (int nt = 0; nt < 2; ++nt) {
        int kv = kv0 + nt * 16 + n16;
#pragma unroll
        for (int r = 0; r < 8; ++r)
          if (kv > q0 + r + 8 * hi) sc[nt].f[r] = -3.0e38f;
      }

      // ---- online softmax ----
      float corr[8];
#pragma unroll
      for (int r = 0; r < 8; ++r) {
        float v = fmaxf(sc[0].f[r], sc[1].f[r]);
#pragma unroll
        for (int off = 8; off >= 1; off >>= 1) v = fmaxf(v, __shfl_xor(v, off, 32));
        float Mn = fmaxf(M[r], v);
        corr[r] = __expf(M[r] - Mn);
        M[r] = Mn;
      }
#pragma unroll
      for (int r = 0; r < 8; ++r) {
        float p0 = __expf(sc[0].f[r] - M[r]);
        float p1 = __expf(sc[1].f[r] - M[r]);
        sc[0].f[r] = p0; sc[1].f[r] = p1;
        float v = p0 + p1;
#pragma unroll
        for (int off = 8; off >= 1; off >>= 1) v += __shfl_xor(v, off, 32);
        L[r] = L[r] * corr[r] + v;
      }
#pragma unroll
      for (int ct = 0; ct < 8; ++ct)
#pragma unroll
        for (int r = 0; r < 8; ++r) O[ct].f[r] *= corr[r];

      // ---- P (C layout) -> LDS -> A fragment ----
#pragma unroll
      for (int nt = 0; nt < 2; ++nt)
#pragma unroll
        for (int r = 0; r < 8; ++r)
          myP[(r + 8 * hi) * 36 + nt * 16 + n16] = f2bf(sc[nt].f[r]);
      asm volatile("s_wait_dscnt 0" ::: "memory");

      BfFrag ap;
#pragma unroll
      for (int j = 0; j < 8; ++j)
        ap.u[j] = *(const unsigned*)(myP + n16 * 36 + a_k2(j, hi));

      // ---- O += P V ----
#pragma unroll
      for (int ct = 0; ct < 8; ++ct) {
        BfFrag bv;
#pragma unroll
        for (int j = 0; j < 8; ++j)
          bv.u[j] = *(const unsigned*)(V + (ct * 16 + n16) * LDV + a_k2(j, hi));
        O[ct].v = __builtin_amdgcn_wmma_f32_16x16x32_bf16(false, ap.v, false, bv.v,
                                                          (short)0, O[ct].v, false, false);
      }
    }

    wait_async0();
    __syncthreads();
  }

  // ---- normalize and store (bf16, consumed by the output projection) ----
#pragma unroll
  for (int ct = 0; ct < 8; ++ct)
#pragma unroll
    for (int r = 0; r < 8; ++r)
      Ao[(size_t)(q0 + r + 8 * hi) * (kNH * kHD) + h * kHD + ct * 16 + n16] =
          f2bf(O[ct].f[r] * (1.0f / L[r]));
}

}  // namespace gqa

extern "C" void kernel_launch(void* const* d_in, const int* in_sizes, int n_in,
                              void* d_out, int out_size, void* d_ws, size_t ws_size,
                              hipStream_t stream) {
  using namespace gqa;
  (void)in_sizes; (void)n_in; (void)out_size; (void)ws_size;

  const float* x     = (const float*)d_in[0];
  const float* Wq    = (const float*)d_in[1];
  const float* Wk    = (const float*)d_in[2];
  const float* Wv    = (const float*)d_in[3];
  const float* Wproj = (const float*)d_in[4];
  const float* qg    = (const float*)d_in[5];
  float* out = (float*)d_out;

  // Workspace layout (~72 MB total)
  char* ws = (char*)d_ws;
  unsigned short* xb  = (unsigned short*)ws; ws += (size_t)kS * kD * 2;
  unsigned short* Wqb = (unsigned short*)ws; ws += (size_t)(kNH * kHD) * kD * 2;
  unsigned short* Wkb = (unsigned short*)ws; ws += (size_t)(kNKV * kHD) * kD * 2;
  unsigned short* Wvb = (unsigned short*)ws; ws += (size_t)(kNKV * kHD) * kD * 2;
  unsigned short* Wpb = (unsigned short*)ws; ws += (size_t)kD * (kNH * kHD) * 2;
  float* Qf = (float*)ws;                    ws += (size_t)kS * (kNH * kHD) * 4;
  float* Kf = (float*)ws;                    ws += (size_t)kS * (kNKV * kHD) * 4;
  float* Vf = (float*)ws;                    ws += (size_t)kS * (kNKV * kHD) * 4;
  unsigned short* Qb  = (unsigned short*)ws; ws += (size_t)kS * (kNH * kHD) * 2;
  unsigned short* Kb  = (unsigned short*)ws; ws += (size_t)kS * (kNKV * kHD) * 2;
  unsigned short* Vt  = (unsigned short*)ws; ws += (size_t)kS * (kNKV * kHD) * 2;
  unsigned short* Aob = (unsigned short*)ws;

  dim3 blk128(128), blk256(256);
  // Operand conversion f32 -> bf16
  to_bf16<<<dim3((kS * kD) / 2048), blk256, 0, stream>>>(x, xb);
  to_bf16<<<dim3(((kNH * kHD) * kD) / 2048), blk256, 0, stream>>>(Wq, Wqb);
  to_bf16<<<dim3(((kNKV * kHD) * kD) / 2048), blk256, 0, stream>>>(Wk, Wkb);
  to_bf16<<<dim3(((kNKV * kHD) * kD) / 2048), blk256, 0, stream>>>(Wv, Wvb);
  to_bf16<<<dim3((kD * (kNH * kHD)) / 2048), blk256, 0, stream>>>(Wproj, Wpb);
  // QKV projections
  gemm_bf16<<<dim3((kNH * kHD) / 64, kS / 128), blk256, 0, stream>>>(xb, Wqb, Qf, kNH * kHD, kD);
  gemm_bf16<<<dim3((kNKV * kHD) / 64, kS / 128), blk256, 0, stream>>>(xb, Wkb, Kf, kNKV * kHD, kD);
  gemm_bf16<<<dim3((kNKV * kHD) / 64, kS / 128), blk256, 0, stream>>>(xb, Wvb, Vf, kNKV * kHD, kD);
  // RMSNorm + RoPE (+ gain and 1/sqrt(HD) folded into Q)
  normrope<<<dim3(kS * kNH / 4), blk128, 0, stream>>>(Qf, qg, Qb, kNH, 0.08838834764831845f);
  normrope<<<dim3(kS * kNKV / 4), blk128, 0, stream>>>(Kf, nullptr, Kb, kNKV, 1.0f);
  // V -> bf16 transposed
  vtrans<<<dim3((kNKV * kHD) / 32, kS / 32), blk256, 0, stream>>>(Vf, Vt);
  // Flash attention
  flash_attn<<<dim3(kS / 64, kNH), blk128, 0, stream>>>(Qb, Kb, Vt, Aob);
  // Output projection
  gemm_bf16<<<dim3(kD / 64, kS / 128), blk256, 0, stream>>>(Aob, Wpb, out, kD, kNH * kHD);
}
